// DecoderRNN_58866821759508
// MI455X (gfx1250) — matrix operand order repaired and checked by reading, not compiled, not run on hardware
//
#include <hip/hip_runtime.h>
#include <cstddef>

typedef __attribute__((ext_vector_type(16))) __bf16 bf16x16;
typedef __attribute__((ext_vector_type(8)))  __bf16 bf16x8;
typedef __attribute__((ext_vector_type(8)))  float  f32x8;
typedef __attribute__((ext_vector_type(4)))  unsigned int u32x4;
typedef __attribute__((ext_vector_type(8)))  int i32x8;
typedef __attribute__((ext_vector_type(4)))  int i32x4;

#define B_SZ   4096
#define ENC    2048
#define DEC    1024
#define G4     4096      // 4*DEC
#define EMB    128
#define S_STEPS 10
#define NT     4
#define NM     11
#define START_TOK 15

// ---------------- elementwise f32 -> bf16 ----------------
__global__ void k_conv_bf16(const float* __restrict__ src, __bf16* __restrict__ dst, int n) {
  int i = blockIdx.x * blockDim.x + threadIdx.x;
  if (i < n) dst[i] = (__bf16)src[i];
}

// dst[r*dst_cols + k] = src[r*src_stride + k]
__global__ void k_conv_bf16_rows(const float* __restrict__ src, __bf16* __restrict__ dst,
                                 int rows, int src_stride, int dst_cols) {
  int i = blockIdx.x * blockDim.x + threadIdx.x;
  int total = rows * dst_cols;
  if (i >= total) return;
  int r = i / dst_cols, k = i - r * dst_cols;
  dst[i] = (__bf16)src[(size_t)r * src_stride + k];
}

// base[g] = b_ih[g] + b_hh[g] + sum_e W_ih[g][ENC+e] * branch_emb0[e]
__global__ void k_base(const float* __restrict__ W_ih, const float* __restrict__ b_ih,
                       const float* __restrict__ b_hh, const float* __restrict__ bemb,
                       float* __restrict__ base) {
  int g = blockIdx.x * blockDim.x + threadIdx.x;
  if (g >= G4) return;
  const float* row = W_ih + (size_t)g * (ENC + 2 * EMB) + ENC;
  float s = b_ih[g] + b_hh[g];
  #pragma unroll 8
  for (int e = 0; e < EMB; ++e) s += row[e] * bemb[e];
  base[g] = s;
}

// table[a][g] = sum_e W_ih[g][ENC+EMB+e] * action_emb[a][e]
__global__ void k_table(const float* __restrict__ W_ih, const float* __restrict__ aemb,
                        float* __restrict__ table) {
  int i = blockIdx.x * blockDim.x + threadIdx.x;
  if (i >= 16 * G4) return;
  int a = i >> 12, g = i & (G4 - 1);
  const float* row = W_ih + (size_t)g * (ENC + 2 * EMB) + ENC + EMB;
  const float* em  = aemb + (size_t)a * EMB;
  float s = 0.f;
  #pragma unroll 8
  for (int e = 0; e < EMB; ++e) s += row[e] * em[e];
  table[i] = s;
}

// ---------------- TDM issue: 2D/3D bf16 tile -> LDS ----------------
// D# group0: count=1, lds_addr, 57-bit global addr, type=2.
// D# group1: data_size=2B; huge tensor dims (no clipping); tile dims; strides (elements).
__device__ __forceinline__ void tdm_issue(const void* gptr, unsigned lds_addr,
                                          int tile_d0, int tile_d1, int tile_d2,
                                          unsigned long long stride0,
                                          unsigned long long stride1) {
  unsigned long long ga = (unsigned long long)gptr;
  u32x4 g0;
  g0.x = 1u;                                              // count=1, user mode
  g0.y = lds_addr;
  g0.z = (unsigned)(ga & 0xFFFFFFFFu);
  g0.w = (unsigned)((ga >> 32) & 0x1FFFFFFu) | (2u << 30); // addr[56:32], type=2
  const unsigned td0 = 1u << 30, td1 = 1u << 30;
  i32x8 g1;
  g1[0] = (int)(1u << 16);                                 // data_size=1 (2 bytes)
  g1[1] = (int)((td0 & 0xFFFFu) << 16);
  g1[2] = (int)(((td0 >> 16) & 0xFFFFu) | ((td1 & 0xFFFFu) << 16));
  g1[3] = (int)(((td1 >> 16) & 0xFFFFu) | ((unsigned)tile_d0 << 16));
  g1[4] = (int)((unsigned)tile_d1 | ((unsigned)tile_d2 << 16));
  g1[5] = (int)(stride0 & 0xFFFFFFFFu);
  g1[6] = (int)(((stride0 >> 32) & 0xFFFFu) | ((unsigned)(stride1 & 0xFFFFu) << 16));
  g1[7] = (int)((stride1 >> 16) & 0xFFFFFFFFu);
  i32x4 z4 = {0, 0, 0, 0};
#if __has_include(<hip/amd_detail/amd_gfx1250_TDM.h>)
  i32x8 z8 = {0, 0, 0, 0, 0, 0, 0, 0};
  __builtin_amdgcn_tensor_load_to_lds(g0, g1, z4, z4, z8, 0);
#else
  __builtin_amdgcn_tensor_load_to_lds(g0, g1, z4, z4, 0);
#endif
}

__device__ __forceinline__ float sigf(float x) { return 1.f / (1.f + __expf(-x)); }

// ---------------- unified WMMA GEMM (TDM + LDS double-buffered) ----------------
// D[4096 x 256cols-per-block] = A[4096 x K] * Bw[rows x K]^T.
// Block output cols organized as 4 chunks of 64, chunk q = cb + q*ROWJUMP + (0..63).
//   zpre  (ROWJUMP=64,  FUSED=0): cb = blockIdx.x*256, contiguous cols; D += bias.
//   step  (ROWJUMP=1024,FUSED=1): cb = blockIdx.x*64 = d-base; chunk q = gate q.
//       Epilogue fuses Zgate + action-table adds and the full LSTM pointwise update.
template <int ROWJUMP, bool FUSED>
__global__ __launch_bounds__(256) void k_gemm(
    const __bf16* __restrict__ A, const __bf16* __restrict__ Bw, int K,
    float* __restrict__ D, const float* __restrict__ bias,
    const float* __restrict__ Zg, const float* __restrict__ table,
    const int* __restrict__ prev, int prev_stride,
    float* __restrict__ c, float* __restrict__ h, __bf16* __restrict__ hb_out)
{
  // LDS staging: per stage (CHUNK=64 K-elems): A 64x128B = 8KB, B 256x128B = 32KB.
  __shared__ alignas(1024) unsigned char smem[2 * (8192 + 32768)];
  const unsigned lds0 = (unsigned)(unsigned long long)(const void*)&smem[0];

  const int lane  = threadIdx.x & 31;
  const int wave  = threadIdx.x >> 5;
  const int half  = lane >> 4;
  const int l16   = lane & 15;
  const int waveM = wave >> 2;      // 0..1
  const int waveN = wave & 3;       // 0..3
  const int blockM = blockIdx.y * 64;
  const int cb = blockIdx.x * (FUSED ? 64 : 256);

  f32x8 acc[2][4] = {};

  const int nStages = K >> 6;       // CHUNK = 64
  if (wave == 0) {   // stage 0 TDM issue (EXEC ignored by TDM; one wave issues)
    tdm_issue(A + (size_t)blockM * K, lds0, 64, 64, 0, (unsigned long long)K, 0ull);
    tdm_issue(Bw + (size_t)cb * K, lds0 + 16384, 64, 64, 4,
              (unsigned long long)K, (unsigned long long)ROWJUMP * (unsigned)K);
  }

  for (int st = 0; st < nStages; ++st) {
    if (wave == 0) {
      if (st + 1 < nStages) {
        const unsigned buf = (unsigned)((st + 1) & 1);
        tdm_issue(A + (size_t)blockM * K + (st + 1) * 64, lds0 + buf * 8192,
                  64, 64, 0, (unsigned long long)K, 0ull);
        tdm_issue(Bw + (size_t)cb * K + (st + 1) * 64, lds0 + 16384 + buf * 32768,
                  64, 64, 4, (unsigned long long)K,
                  (unsigned long long)ROWJUMP * (unsigned)K);
        __builtin_amdgcn_s_wait_tensorcnt(2);   // stage st complete (in-order)
      } else {
        __builtin_amdgcn_s_wait_tensorcnt(0);
      }
    }
    __syncthreads();

    const unsigned char* Ab = &smem[(st & 1) * 8192];
    const unsigned char* Bb = &smem[16384 + (st & 1) * 32768];
    #pragma unroll
    for (int ks = 0; ks < 2; ++ks) {           // two 32-K substeps per stage
      // A fragments (ISA 16-bit A 16x32 layout; rows are 128B in LDS)
      union { bf16x16 v; bf16x8 h2[2]; } a0, a1;
      const unsigned char* ar0 = Ab + (size_t)(waveM * 32 + l16) * 128 + ks * 64;
      a0.h2[0] = *(const bf16x8*)(ar0 + half * 16);
      a0.h2[1] = *(const bf16x8*)(ar0 + 32 + half * 16);
      const unsigned char* ar1 = ar0 + 16 * 128;
      a1.h2[0] = *(const bf16x8*)(ar1 + half * 16);
      a1.h2[1] = *(const bf16x8*)(ar1 + 32 + half * 16);
      // B fragments (ISA 16-bit B 32x16 layout): two 16B chunks per lane, matching A's k order
      union { bf16x16 v; bf16x8 h2[2]; } b0, b1, b2, b3;
      {
        const unsigned char* brb = Bb + (size_t)(waveN * 16 + l16) * 128 + ks * 64 + half * 16;
        b0.h2[0] = *(const bf16x8*)(brb);
        b0.h2[1] = *(const bf16x8*)(brb + 32);
        b1.h2[0] = *(const bf16x8*)(brb + 64 * 128);
        b1.h2[1] = *(const bf16x8*)(brb + 64 * 128 + 32);
        b2.h2[0] = *(const bf16x8*)(brb + 128 * 128);
        b2.h2[1] = *(const bf16x8*)(brb + 128 * 128 + 32);
        b3.h2[0] = *(const bf16x8*)(brb + 192 * 128);
        b3.h2[1] = *(const bf16x8*)(brb + 192 * 128 + 32);
      }
      // Keep the 12 ds_load_b128 of this substep grouped ahead of the 8 WMMAs:
      // distinct fragment regs -> one DScnt wait, WMMA/LDS co-execution.
      __builtin_amdgcn_sched_barrier(0);
      acc[0][0] = __builtin_amdgcn_wmma_f32_16x16x32_bf16(
          false, a0.v, false, b0.v, (short)0, acc[0][0], false, false);
      acc[1][0] = __builtin_amdgcn_wmma_f32_16x16x32_bf16(
          false, a1.v, false, b0.v, (short)0, acc[1][0], false, false);
      acc[0][1] = __builtin_amdgcn_wmma_f32_16x16x32_bf16(
          false, a0.v, false, b1.v, (short)0, acc[0][1], false, false);
      acc[1][1] = __builtin_amdgcn_wmma_f32_16x16x32_bf16(
          false, a1.v, false, b1.v, (short)0, acc[1][1], false, false);
      acc[0][2] = __builtin_amdgcn_wmma_f32_16x16x32_bf16(
          false, a0.v, false, b2.v, (short)0, acc[0][2], false, false);
      acc[1][2] = __builtin_amdgcn_wmma_f32_16x16x32_bf16(
          false, a1.v, false, b2.v, (short)0, acc[1][2], false, false);
      acc[0][3] = __builtin_amdgcn_wmma_f32_16x16x32_bf16(
          false, a0.v, false, b3.v, (short)0, acc[0][3], false, false);
      acc[1][3] = __builtin_amdgcn_wmma_f32_16x16x32_bf16(
          false, a1.v, false, b3.v, (short)0, acc[1][3], false, false);
    }
    __syncthreads();   // all waves done reading before buf is overwritten
  }

  // Epilogue. C/D layout: VGPR r -> lanes 0-15 M=r, lanes 16-31 M=8+r; N=l16.
  #pragma unroll
  for (int mt = 0; mt < 2; ++mt) {
    #pragma unroll
    for (int r = 0; r < 8; ++r) {
      const int row = blockM + waveM * 32 + mt * 16 + half * 8 + r;
      if (FUSED) {
        const int d = cb + waveN * 16 + l16;
        const int p = prev ? prev[(size_t)row * prev_stride] : START_TOK;
        const float* tb = table + (size_t)p * G4;
        const float* zr = Zg + (size_t)row * G4;
        float gi = acc[mt][0][r] + zr[d]            + tb[d];
        float gf = acc[mt][1][r] + zr[DEC + d]      + tb[DEC + d];
        float gg = acc[mt][2][r] + zr[2 * DEC + d]  + tb[2 * DEC + d];
        float go = acc[mt][3][r] + zr[3 * DEC + d]  + tb[3 * DEC + d];
        const size_t idx = (size_t)row * DEC + d;
        float cn = sigf(gf) * c[idx] + sigf(gi) * tanhf(gg);
        float hn = sigf(go) * tanhf(cn);
        c[idx] = cn;
        h[idx] = hn;
        hb_out[idx] = (__bf16)hn;
      } else {
        float* drow = D + (size_t)row * G4;
        #pragma unroll
        for (int nt = 0; nt < 4; ++nt) {
          const int col = cb + nt * ROWJUMP + waveN * 16 + l16;
          drow[col] = acc[mt][nt][r] + bias[col];
        }
      }
    }
  }
}

// ---------------- tiny logits + log-softmax gather (one wave per batch row) ----
template <int C>
__global__ void k_logits(const float* __restrict__ h, const float* __restrict__ Wc,
                         const float* __restrict__ bc,
                         const int* __restrict__ act, int act_stride,
                         float* __restrict__ out, int out_stride) {
  int gwave = (blockIdx.x * blockDim.x + threadIdx.x) >> 5;
  int lane  = threadIdx.x & 31;
  if (gwave >= B_SZ) return;
  const float* hr = h + (size_t)gwave * DEC;
  float acc[C];
  #pragma unroll
  for (int ci = 0; ci < C; ++ci) acc[ci] = 0.f;
  for (int k = lane; k < DEC; k += 32) {
    float hv = hr[k];
    #pragma unroll
    for (int ci = 0; ci < C; ++ci) acc[ci] += hv * Wc[(size_t)ci * DEC + k];
  }
  #pragma unroll
  for (int off = 16; off > 0; off >>= 1)
    #pragma unroll
    for (int ci = 0; ci < C; ++ci) acc[ci] += __shfl_down(acc[ci], off, 32);
  if (lane == 0) {
    float m = -1e30f;
    #pragma unroll
    for (int ci = 0; ci < C; ++ci) { acc[ci] += bc[ci]; m = fmaxf(m, acc[ci]); }
    float s = 0.f;
    #pragma unroll
    for (int ci = 0; ci < C; ++ci) s += __expf(acc[ci] - m);
    float lse = m + __logf(s);
    int a = act[(size_t)gwave * act_stride];
    out[(size_t)gwave * out_stride] = acc[a] - lse;
  }
}

// ---------------- write replayed integer actions as floats ----------------
__global__ void k_actions(const int* __restrict__ t, const int* __restrict__ m,
                          float* __restrict__ out) {
  int i = blockIdx.x * blockDim.x + threadIdx.x;
  const int n = B_SZ * 2 * S_STEPS;
  if (i >= n) return;
  out[i]         = (float)t[i];
  out[2 * n + i] = (float)m[i];
}

extern "C" void kernel_launch(void* const* d_in, const int* in_sizes, int n_in,
                              void* d_out, int out_size, void* d_ws, size_t ws_size,
                              hipStream_t stream) {
  (void)in_sizes; (void)n_in; (void)out_size; (void)ws_size;
  const float* z1    = (const float*)d_in[0];
  const float* z2    = (const float*)d_in[1];
  const float* aemb  = (const float*)d_in[2];
  const float* bemb  = (const float*)d_in[3];
  const float* W_ih  = (const float*)d_in[4];
  const float* W_hh  = (const float*)d_in[5];
  const float* b_ih  = (const float*)d_in[6];
  const float* b_hh  = (const float*)d_in[7];
  const float* Wt    = (const float*)d_in[8];
  const float* bt    = (const float*)d_in[9];
  const float* Wm    = (const float*)d_in[10];
  const float* bm    = (const float*)d_in[11];
  const int*   old_t = (const int*)d_in[12];
  const int*   old_m = (const int*)d_in[13];

  // ---- workspace layout
  char* p = (char*)d_ws;
  size_t o = 0;
  auto take = [&](size_t bytes) { void* r = p + o; o = (o + bytes + 255) & ~(size_t)255; return r; };
  __bf16* whh_bf  = (__bf16*)take((size_t)G4 * DEC * 2);           // 8 MB
  __bf16* wihz_bf = (__bf16*)take((size_t)G4 * ENC * 2);           // 16 MB
  __bf16* zb0     = (__bf16*)take((size_t)B_SZ * ENC * 2);         // 16 MB
  __bf16* zb1     = (__bf16*)take((size_t)B_SZ * ENC * 2);         // 16 MB
  float*  base    = (float*)take((size_t)G4 * 4);
  float*  table   = (float*)take((size_t)16 * G4 * 4);
  float*  Zgate   = (float*)take((size_t)2 * B_SZ * G4 * 4);       // 128 MB
  float*  h       = (float*)take((size_t)B_SZ * DEC * 4);          // 16 MB
  float*  c       = (float*)take((size_t)B_SZ * DEC * 4);          // 16 MB
  __bf16* hb0     = (__bf16*)take((size_t)B_SZ * DEC * 2);         // 8 MB
  __bf16* hb1     = (__bf16*)take((size_t)B_SZ * DEC * 2);         // 8 MB

  const int TB = 256;
  // ---- precompute: conversions
  k_conv_bf16<<<(G4 * DEC + TB - 1) / TB, TB, 0, stream>>>(W_hh, whh_bf, G4 * DEC);
  k_conv_bf16_rows<<<(G4 * ENC + TB - 1) / TB, TB, 0, stream>>>(W_ih, wihz_bf, G4, ENC + 2 * EMB, ENC);
  k_conv_bf16<<<(B_SZ * ENC + TB - 1) / TB, TB, 0, stream>>>(z1, zb0, B_SZ * ENC);
  k_conv_bf16<<<(B_SZ * ENC + TB - 1) / TB, TB, 0, stream>>>(z2, zb1, B_SZ * ENC);
  // ---- precompute: bias-vector and action table
  k_base<<<G4 / TB, TB, 0, stream>>>(W_ih, b_ih, b_hh, bemb, base);
  k_table<<<16 * G4 / TB, TB, 0, stream>>>(W_ih, aemb, table);
  // ---- precompute: per-branch Zgate = z @ W_ih[:, :ENC].T + base
  dim3 ggrid(16, 64);
  k_gemm<64, false><<<ggrid, TB, 0, stream>>>(zb0, wihz_bf, ENC, Zgate, base,
                                              nullptr, nullptr, nullptr, 0,
                                              nullptr, nullptr, nullptr);
  k_gemm<64, false><<<ggrid, TB, 0, stream>>>(zb1, wihz_bf, ENC,
                                              Zgate + (size_t)B_SZ * G4, base,
                                              nullptr, nullptr, nullptr, 0,
                                              nullptr, nullptr, nullptr);

  // ---- init recurrent state
  hipMemsetAsync(c, 0, (size_t)B_SZ * DEC * 4, stream);
  hipMemsetAsync(hb0, 0, (size_t)B_SZ * DEC * 2, stream);

  const int N20 = B_SZ * 2 * S_STEPS;
  float* out_f   = (float*)d_out;
  float* out_tlp = out_f + N20;
  float* out_mlp = out_f + 3 * N20;

  const int log_blocks = (B_SZ * 32 + TB - 1) / TB;
  __bf16* hbuf[2] = {hb0, hb1};
  int cur = 0;

  const int* prevPtr = nullptr;   // start token for the very first step
  for (int br = 0; br < 2; ++br) {
    const float* Zg = Zgate + (size_t)br * B_SZ * G4;
    for (int s = 0; s < S_STEPS; ++s) {
      // --- sub-step 0: transform head (fused GEMM + LSTM update)
      k_gemm<DEC, true><<<ggrid, TB, 0, stream>>>(hbuf[cur], whh_bf, DEC,
                                                  nullptr, nullptr, Zg, table,
                                                  prevPtr, 2 * S_STEPS,
                                                  c, h, hbuf[cur ^ 1]);
      cur ^= 1;
      k_logits<NT><<<log_blocks, TB, 0, stream>>>(h, Wt, bt,
                                                  old_t + br * S_STEPS + s, 2 * S_STEPS,
                                                  out_tlp + br * S_STEPS + s, 2 * S_STEPS);
      prevPtr = old_t + br * S_STEPS + s;
      // --- sub-step 1: magnitude head
      k_gemm<DEC, true><<<ggrid, TB, 0, stream>>>(hbuf[cur], whh_bf, DEC,
                                                  nullptr, nullptr, Zg, table,
                                                  prevPtr, 2 * S_STEPS,
                                                  c, h, hbuf[cur ^ 1]);
      cur ^= 1;
      k_logits<NM><<<log_blocks, TB, 0, stream>>>(h, Wm, bm,
                                                  old_m + br * S_STEPS + s, 2 * S_STEPS,
                                                  out_mlp + br * S_STEPS + s, 2 * S_STEPS);
      prevPtr = old_m + br * S_STEPS + s;
    }
  }

  // ---- replayed actions -> output sections 0 and 2
  k_actions<<<(N20 + TB - 1) / TB, TB, 0, stream>>>(old_t, old_m, out_f);
}
